// Multi_head_attention_23888608101213
// MI455X (gfx1250) — compile-verified, matrix operand order checked by reading, not run on hardware
//
#include <hip/hip_runtime.h>
#include <hip/hip_bf16.h>

typedef __attribute__((ext_vector_type(16))) _Float16 v16h;
typedef __attribute__((ext_vector_type(8)))  _Float16 v8h;
typedef __attribute__((ext_vector_type(8)))  float    v8f;

#define BB 2
#define HH 8
#define LL 2048
#define DD 512
#define HD 64

// Load a 16x32 f16 fragment row-slice: 32 contiguous halves from one logical
// row, split per ISA layout (lane-half hi selects K chunks [hi*8, hi*8+8) and
// [16+hi*8, 16+hi*8+8)).
__device__ __forceinline__ v16h frag_load(const _Float16* rowptr, int hi) {
    union { v16h v; v8h h[2]; } u;
    u.h[0] = *(const v8h*)(rowptr + hi * 8);
    u.h[1] = *(const v8h*)(rowptr + 16 + hi * 8);
    return u.v;
}

__device__ __forceinline__ v8f wmma16(v16h a, v16h b, v8f c) {
    return __builtin_amdgcn_wmma_f32_16x16x32_f16(
        /*neg_a=*/false, a, /*neg_b=*/false, b,
        /*c_mod=*/(short)0, c, /*reuse_a=*/false, /*reuse_b=*/false);
}

__global__ void cvt_f32_to_f16(const float* __restrict__ src,
                               _Float16* __restrict__ dst, int n) {
    int i = blockIdx.x * blockDim.x + threadIdx.x;
    if (i < n) dst[i] = (_Float16)src[i];
}

// Streaming kernel for the strictly-future region not covered by the attn
// kernel's compute blocks: score = prev - 1e9. Fully vectorized b128 traffic;
// this is ~half of the dominant 536MB HBM stream.
__global__ void mask_fill_kernel(const float* __restrict__ prev,
                                 float* __restrict__ score_out) {
    const int qr = blockIdx.x;           // 0..2047
    const int bh = blockIdx.y;           // 0..15
    const int q0 = qr & ~15;
    const int cstart = (((q0 + 15) >> 5) + 1) << 5;   // first col not written by attn
    const float4* p4 = (const float4*)(prev + ((size_t)bh * LL + qr) * LL);
    float4* s4 = (float4*)(score_out + ((size_t)bh * LL + qr) * LL);
    for (int c4 = (cstart >> 2) + threadIdx.x; c4 < (LL >> 2); c4 += blockDim.x) {
        float4 f = p4[c4];
        f.x -= 1e9f; f.y -= 1e9f; f.z -= 1e9f; f.w -= 1e9f;
        s4[c4] = f;
    }
}

// One wave computes a 16x64 tile of Y = X @ W^T + b, applies RoPE (q,k),
// stores f16: q/k as [bh][L][hd], v transposed as [bh][hd][L].
__global__ void proj_rope_kernel(const _Float16* __restrict__ xh,
                                 const _Float16* __restrict__ wh,  // 3 mats concat
                                 const float* __restrict__ bq,
                                 const float* __restrict__ bk,
                                 const float* __restrict__ bv,
                                 _Float16* __restrict__ q16,
                                 _Float16* __restrict__ k16,
                                 _Float16* __restrict__ vT16) {
    const int lane = threadIdx.x & 31;
    const int lt = lane & 15, hi = lane >> 4;
    const int wid = (blockIdx.x * blockDim.x + threadIdx.x) >> 5;
    const int mat   = wid / (256 * 8);   // 0=q 1=k 2=v
    const int rem   = wid % (256 * 8);
    const int mtile = rem >> 3;          // 0..255 (16 rows of 4096)
    const int ngrp  = rem & 7;           // head / 64-col group

    const _Float16* wmat = wh + (size_t)mat * (DD * DD);
    const float* bias = (mat == 0) ? bq : (mat == 1) ? bk : bv;

    v8f acc[4];
#pragma unroll
    for (int nn = 0; nn < 4; ++nn)
        acc[nn] = (v8f){0.f,0.f,0.f,0.f,0.f,0.f,0.f,0.f};

    const _Float16* arow = xh + (size_t)(mtile * 16 + lt) * DD;
    for (int kk = 0; kk < DD; kk += 32) {
        v16h a = frag_load(arow + kk, hi);
#pragma unroll
        for (int nn = 0; nn < 4; ++nn) {
            const _Float16* brow = wmat + (size_t)(ngrp * 64 + nn * 16 + lt) * DD;
            v16h b = frag_load(brow + kk, hi);
            acc[nn] = wmma16(a, b, acc[nn]);
        }
    }

    const int bi = (mtile * 16) / LL;
    const int l0 = (mtile * 16) % LL;
    const int head = ngrp;
#pragma unroll
    for (int nn = 0; nn < 4; ++nn) {
        const int nc = nn * 16 + lt;                 // col within head (0..63)
        const float bsv = bias[ngrp * 64 + nc];
        const int p2 = nc & ~1;
        const float invf = __powf(10000.0f, -(float)p2 / 64.0f);
#pragma unroll
        for (int r = 0; r < 8; ++r) {
            const int rl = r + 8 * hi;               // C layout: M = r + 8*(lane/16)
            const int l = l0 + rl;
            float val = acc[nn][r] + bsv;
            if (mat != 2) {                          // RoPE, interleaved pairs
                float pv = __shfl_xor(val, 1);       // partner column value
                float ang = (float)l * invf;
                float cs = __cosf(ang), sn = __sinf(ang);
                val = val * cs + ((nc & 1) ? pv : -pv) * sn;
            }
            _Float16 hv = (_Float16)val;
            const size_t bh = (size_t)(bi * HH + head);
            if (mat == 0)      q16[(bh * LL + l) * HD + nc] = hv;
            else if (mat == 1) k16[(bh * LL + l) * HD + nc] = hv;
            else               vT16[(bh * HD + nc) * LL + l] = hv;
        }
    }
}

// One wave per (bh, 16-query tile). Single pass over the causal region:
// score = mask(QK^T/8)+prev is written once; online softmax accumulates
// O = softmax(score) @ V via WMMA. Future region handled by mask_fill_kernel.
__global__ void attn_kernel(const _Float16* __restrict__ q16,
                            const _Float16* __restrict__ k16,
                            const _Float16* __restrict__ vT16,
                            const float* __restrict__ prev,
                            float* __restrict__ o_out,
                            float* __restrict__ score_out) {
    __shared__ _Float16 lds[8][16][40];  // per-wave 16x32 P tile, padded
    const int lane = threadIdx.x & 31;
    const int lt = lane & 15, hi = lane >> 4;
    const int w = threadIdx.x >> 5;
    const int wid = (blockIdx.x * blockDim.x + threadIdx.x) >> 5;
    const int bh = wid >> 7;     // 0..15
    const int qi = wid & 127;    // query tile
    const int q0 = qi * 16;

    const _Float16* qb = q16 + (size_t)bh * LL * HD;
    const _Float16* kb_ = k16 + (size_t)bh * LL * HD;
    const _Float16* vb = vT16 + (size_t)bh * HD * LL;
    const float* pb = prev + (size_t)bh * LL * LL;
    float* sb = score_out + (size_t)bh * LL * LL;

    v16h aq0 = frag_load(qb + (size_t)(q0 + lt) * HD + 0, hi);
    v16h aq1 = frag_load(qb + (size_t)(q0 + lt) * HD + 32, hi);

    v8f oacc[4];
#pragma unroll
    for (int nn = 0; nn < 4; ++nn)
        oacc[nn] = (v8f){0.f,0.f,0.f,0.f,0.f,0.f,0.f,0.f};
    float mi[8], li[8];
#pragma unroll
    for (int r = 0; r < 8; ++r) { mi[r] = -3.0e38f; li[r] = 0.f; }

    const float scale = 0.125f;  // 1/sqrt(64)

    for (int kt = 0; kt <= q0 + 15; kt += 32) {
        // ---- S = Q(16x64) @ K^T for two 16-key tiles ----
        v16h bL0 = frag_load(kb_ + (size_t)(kt + lt) * HD + 0, hi);
        v16h bL1 = frag_load(kb_ + (size_t)(kt + lt) * HD + 32, hi);
        v8f sL = (v8f){0.f,0.f,0.f,0.f,0.f,0.f,0.f,0.f};
        sL = wmma16(aq0, bL0, sL);
        sL = wmma16(aq1, bL1, sL);
        v16h bR0 = frag_load(kb_ + (size_t)(kt + 16 + lt) * HD + 0, hi);
        v16h bR1 = frag_load(kb_ + (size_t)(kt + 16 + lt) * HD + 32, hi);
        v8f sR = (v8f){0.f,0.f,0.f,0.f,0.f,0.f,0.f,0.f};
        sR = wmma16(aq0, bR0, sR);
        sR = wmma16(aq1, bR1, sR);

        float corr[8];
#pragma unroll
        for (int r = 0; r < 8; ++r) {
            const int rl = r + 8 * hi;
            const int qr = q0 + rl;
            const int cL = kt + lt, cR = kt + 16 + lt;
            float scL = ((cL <= qr) ? sL[r] * scale : -1e9f) + pb[(size_t)qr * LL + cL];
            float scR = ((cR <= qr) ? sR[r] * scale : -1e9f) + pb[(size_t)qr * LL + cR];
            sb[(size_t)qr * LL + cL] = scL;
            sb[(size_t)qr * LL + cR] = scR;
            // row reductions stay within each 16-lane half (bits 0..3)
            float rmax = fmaxf(scL, scR);
#pragma unroll
            for (int m = 1; m < 16; m <<= 1) rmax = fmaxf(rmax, __shfl_xor(rmax, m));
            float mnew = fmaxf(mi[r], rmax);
            float c = __expf(mi[r] - mnew);
            float pL = __expf(scL - mnew);
            float pR = __expf(scR - mnew);
            float rs = pL + pR;
#pragma unroll
            for (int m = 1; m < 16; m <<= 1) rs += __shfl_xor(rs, m);
            li[r] = li[r] * c + rs;
            mi[r] = mnew;
            corr[r] = c;
            lds[w][rl][lt] = (_Float16)pL;       // C-layout -> row-major LDS
            lds[w][rl][16 + lt] = (_Float16)pR;
        }
#pragma unroll
        for (int nn = 0; nn < 4; ++nn)
#pragma unroll
            for (int r = 0; r < 8; ++r) oacc[nn][r] *= corr[r];

        // Wave-internal transpose bounce through LDS (DS ops in-order per wave)
        asm volatile("s_wait_dscnt 0x0" ::: "memory");
        v16h pa = frag_load(&lds[w][lt][0], hi);  // A fragment: 16 rows x 32 keys
#pragma unroll
        for (int nn = 0; nn < 4; ++nn) {
            v16h bv = frag_load(vb + (size_t)(nn * 16 + lt) * LL + kt, hi);
            oacc[nn] = wmma16(pa, bv, oacc[nn]);
        }
    }

    const int bi = bh >> 3, head = bh & 7;
#pragma unroll
    for (int nn = 0; nn < 4; ++nn)
#pragma unroll
        for (int r = 0; r < 8; ++r) {
            const int rl = r + 8 * hi;
            const int qr = q0 + rl;
            o_out[((size_t)bi * LL + qr) * DD + head * HD + nn * 16 + lt] =
                oacc[nn][r] / li[r];
        }
}

extern "C" void kernel_launch(void* const* d_in, const int* in_sizes, int n_in,
                              void* d_out, int out_size, void* d_ws, size_t ws_size,
                              hipStream_t stream) {
    const float* x    = (const float*)d_in[0];
    const float* prev = (const float*)d_in[1];
    const float* Wq   = (const float*)d_in[2];
    const float* bq   = (const float*)d_in[3];
    const float* Wk   = (const float*)d_in[4];
    const float* bk   = (const float*)d_in[5];
    const float* Wv   = (const float*)d_in[6];
    const float* bv   = (const float*)d_in[7];

    float* o_out = (float*)d_out;
    float* score_out = o_out + (size_t)BB * LL * DD;  // outputs concat: o then score

    const size_t NQKV = (size_t)BB * HH * LL * HD;    // 2^21
    _Float16* q16  = (_Float16*)d_ws;
    _Float16* k16  = q16 + NQKV;
    _Float16* vT16 = k16 + NQKV;
    _Float16* xh   = vT16 + NQKV;                     // 2^21 halves
    _Float16* wh   = xh + (size_t)BB * LL * DD;       // 3 * 512*512 halves

    const int n_x = BB * LL * DD;       // 2,097,152
    const int n_w = DD * DD;            // 262,144
    cvt_f32_to_f16<<<(n_x + 255) / 256, 256, 0, stream>>>(x, xh, n_x);
    cvt_f32_to_f16<<<(n_w + 255) / 256, 256, 0, stream>>>(Wq, wh + 0 * (size_t)n_w, n_w);
    cvt_f32_to_f16<<<(n_w + 255) / 256, 256, 0, stream>>>(Wk, wh + 1 * (size_t)n_w, n_w);
    cvt_f32_to_f16<<<(n_w + 255) / 256, 256, 0, stream>>>(Wv, wh + 2 * (size_t)n_w, n_w);

    // Streaming masked-future fill: one row per block, b128 vectorized.
    mask_fill_kernel<<<dim3(LL, BB * HH), 256, 0, stream>>>(prev, score_out);

    // 3 mats * 256 M-tiles * 8 N-groups = 6144 waves, 8 waves/block
    proj_rope_kernel<<<768, 256, 0, stream>>>(xh, wh, bq, bk, bv, q16, k16, vT16);

    // 16 bh * 128 q-tiles = 2048 waves, 8 waves/block
    attn_kernel<<<256, 256, 0, stream>>>(q16, k16, vT16, prev, o_out, score_out);
}
